// LSTMModel_43739946942821
// MI455X (gfx1250) — matrix operand order repaired and checked by reading, not compiled, not run on hardware
//
#include <hip/hip_runtime.h>

// LSTM: B=4096, T=512, I=1, H=32. Each wave processes 16 batch rows.
// Per step: gates[16,128] = one v_wmma_f32_16x16x32_f16 per 16-gate tile (8 tiles),
// A = h (f16, K=32==H), B = W_hh^T tiles resident in VGPRs, C = x*W_ih + bias.

#define HID   32
#define TLEN  512
#define ROWH  40   // padded LDS row stride in halves (80B -> conflict-free rows)

typedef __attribute__((ext_vector_type(16))) _Float16 v16h;
typedef __attribute__((ext_vector_type(8)))  _Float16 v8h;
typedef __attribute__((ext_vector_type(8)))  float    v8f;

__device__ __forceinline__ float fexp(float x) {
    return __builtin_amdgcn_exp2f(x * 1.44269504088896340736f);  // v_exp_f32
}
__device__ __forceinline__ float fsig(float x) {
    return __builtin_amdgcn_rcpf(1.0f + fexp(-x));               // v_rcp_f32
}
__device__ __forceinline__ float ftanh(float x) {
    return fmaf(2.0f, fsig(2.0f * x), -1.0f);
}

__global__ __launch_bounds__(32)
void lstm_wmma_kernel(const float* __restrict__ x,
                      const float* __restrict__ W_ih,
                      const float* __restrict__ W_hh,
                      const float* __restrict__ b_ih,
                      const float* __restrict__ b_hh,
                      const float* __restrict__ W_fc,
                      const float* __restrict__ b_fc,
                      float* __restrict__ out)
{
    __shared__ __align__(16) _Float16 hbuf[16 * ROWH];  // h[16 rows][32 cols], padded

    const int lane  = threadIdx.x & 31;
    const int halfw = lane >> 4;     // 0: lanes 0-15, 1: lanes 16-31
    const int mrow  = lane & 15;     // row (A layout) / column-in-tile (B,C,D layout)
    const int bb    = blockIdx.x * 16;

    // h0 = 0
    for (int i = lane; i < 16 * ROWH; i += 32) hbuf[i] = (_Float16)0.0f;

    // ---- B matrices: W_hh^T tiles, 32x16 f16, resident in VGPRs ----
    // lane holds column n = 16t + mrow; lanes 0-15 carry K=0..15, lanes 16-31 K=16..31.
    v16h Bm[8];
    #pragma unroll
    for (int t = 0; t < 8; ++t) {
        const int n = t * 16 + mrow;
        const float* wr = W_hh + n * HID + halfw * 16;
        #pragma unroll
        for (int k = 0; k < 16; ++k) Bm[t][k] = (_Float16)wr[k];
    }

    // per-(lane,tile) input-projection constants: n = 16t + mrow
    float wih[8], bsum[8];
    #pragma unroll
    for (int t = 0; t < 8; ++t) {
        const int n = t * 16 + mrow;
        wih[t]  = W_ih[n];
        bsum[t] = b_ih[n] + b_hh[n];
    }

    // cell state: 16 cells per lane -> c[m = r + 8*halfw][j = 16*t01 + mrow]
    float cst[2][8];
    #pragma unroll
    for (int a = 0; a < 2; ++a)
        #pragma unroll
        for (int r = 0; r < 8; ++r) cst[a][r] = 0.0f;

    v16h A = {};  // h0 = 0 in A layout

    for (int t4 = 0; t4 < TLEN / 4; ++t4) {
        // 4 timesteps of x per row (global_load_b128); row r+8*halfw matches D layout
        float4 xq[8];
        #pragma unroll
        for (int r = 0; r < 8; ++r) {
            const float4* xr = (const float4*)(x + (size_t)(bb + r + 8 * halfw) * TLEN);
            xq[r] = xr[t4];
        }

        #pragma unroll
        for (int s = 0; s < 4; ++s) {
            float xv[8];
            #pragma unroll
            for (int r = 0; r < 8; ++r) {
                float4 q = xq[r];
                xv[r] = (s == 0) ? q.x : (s == 1) ? q.y : (s == 2) ? q.z : q.w;
            }

            // gates = x*W_ih + (b_ih+b_hh) + h @ W_hh^T   (8 WMMA tiles)
            v8f D[8];
            #pragma unroll
            for (int t = 0; t < 8; ++t) {
                v8f C;
                #pragma unroll
                for (int r = 0; r < 8; ++r) C[r] = fmaf(xv[r], wih[t], bsum[t]);
                D[t] = __builtin_amdgcn_wmma_f32_16x16x32_f16(
                    /*neg_a=*/false, A, /*neg_b=*/false, Bm[t],
                    /*c_mod=*/(short)0, C, /*reuse_a=*/false, /*reuse_b=*/false);
            }

            // elementwise LSTM cell; tiles 0-1=i, 2-3=f, 4-5=g, 6-7=o (PyTorch order)
            #pragma unroll
            for (int t01 = 0; t01 < 2; ++t01) {
                #pragma unroll
                for (int r = 0; r < 8; ++r) {
                    const float ig = fsig (D[t01 + 0][r]);
                    const float fg = fsig (D[t01 + 2][r]);
                    const float gg = ftanh(D[t01 + 4][r]);
                    const float og = fsig (D[t01 + 6][r]);
                    const float c  = fmaf(fg, cst[t01][r], ig * gg);
                    cst[t01][r] = c;
                    const float h = og * ftanh(c);
                    const int m = r + 8 * halfw;
                    const int j = 16 * t01 + mrow;
                    hbuf[m * ROWH + j] = (_Float16)h;   // ds_store_b16
                }
            }

            // reload A (contiguous k order, as in Bm): lane -> row mrow, K in {16*halfw .. 16*halfw+15}
            const _Float16* hp = &hbuf[mrow * ROWH + 16 * halfw];
            v8h lo = *(const v8h*)(hp);        // ds_load_b128
            v8h hi = *(const v8h*)(hp + 8);    // ds_load_b128
            #pragma unroll
            for (int k = 0; k < 8; ++k) { A[k] = lo[k]; A[k + 8] = hi[k]; }
        }
    }

    // final: out[b] = sigmoid(h_last . W_fc + b_fc), one row per lane (lanes 0-15)
    if (lane < 16) {
        float acc = b_fc[0];
        #pragma unroll
        for (int j = 0; j < HID; ++j)
            acc = fmaf((float)hbuf[mrow * ROWH + j], W_fc[j], acc);
        out[bb + mrow] = fsig(acc);
    }
}

extern "C" void kernel_launch(void* const* d_in, const int* in_sizes, int n_in,
                              void* d_out, int out_size, void* d_ws, size_t ws_size,
                              hipStream_t stream) {
    (void)in_sizes; (void)n_in; (void)d_ws; (void)ws_size;
    const float* x    = (const float*)d_in[0];
    const float* W_ih = (const float*)d_in[1];
    const float* W_hh = (const float*)d_in[2];
    const float* b_ih = (const float*)d_in[3];
    const float* b_hh = (const float*)d_in[4];
    const float* W_fc = (const float*)d_in[5];
    const float* b_fc = (const float*)d_in[6];
    float* out = (float*)d_out;

    const int B = out_size;                 // 4096
    dim3 grid(B / 16), block(32);           // one wave per block, 16 batch rows per wave
    hipLaunchKernelGGL(lstm_wmma_kernel, grid, block, 0, stream,
                       x, W_ih, W_hh, b_ih, b_hh, W_fc, b_fc, out);
}
